// NearestNeighborMatcher_89687507075082
// MI455X (gfx1250) — matrix operand order repaired and checked by reading, compiled, not run on hardware
//
#include <hip/hip_runtime.h>
#include <math.h>

// Problem constants (match reference)
#define BB 8
#define NN 4096
#define MM 4096
#define DD 128

#define TILE 128
#define LDH 144   // LDS row stride in halfs (=72 dwords, ==8 mod 64 banks -> conflict-free frags)
#define LDC 133   // C-tile LDS row stride in dwords (bank stride 5 -> conflict-free scans)

typedef _Float16 f16;
typedef __attribute__((ext_vector_type(4)))  _Float16 v4h;
typedef __attribute__((ext_vector_type(8)))  _Float16 v8h;
typedef __attribute__((ext_vector_type(16))) _Float16 v16h;
typedef __attribute__((ext_vector_type(8)))  float    v8f;

struct Part { float v0, v1; int i0, i1; };  // 16 B

// ---------------------------------------------------------------------------
// L2-normalize rows of [rows,128] f32 and split each value into f16 hi + lo:
// x = hi + lo, hi = f16(x), lo = f16(x - hi)  (x - hi is exact in f32).
// One wave per row, float4 per lane.
// ---------------------------------------------------------------------------
__global__ __launch_bounds__(256) void l2norm_split_kernel(
    const float* __restrict__ in, f16* __restrict__ hi, f16* __restrict__ lo, int rows) {
  const int wid = threadIdx.x >> 5, lane = threadIdx.x & 31;
  const int row = blockIdx.x * 8 + wid;
  if (row >= rows) return;
  float4 v = ((const float4*)(in + (size_t)row * DD))[lane];
  float ss = v.x * v.x + v.y * v.y + v.z * v.z + v.w * v.w;
  #pragma unroll
  for (int off = 16; off > 0; off >>= 1) ss += __shfl_xor(ss, off, 32);
  const float s = 1.0f / fmaxf(sqrtf(ss), 1e-12f);
  float x[4] = { v.x * s, v.y * s, v.z * s, v.w * s };
  v4h hv, lv;
  #pragma unroll
  for (int j = 0; j < 4; ++j) {
    f16 h = (f16)x[j];
    hv[j] = h;
    lv[j] = (f16)(x[j] - (float)h);
  }
  *(v4h*)(hi + (size_t)row * DD + lane * 4) = hv;
  *(v4h*)(lo + (size_t)row * DD + lane * 4) = lv;
}

// ---------------------------------------------------------------------------
// Fused split-precision GEMM (f16 WMMA, f32 accumulate) + per-tile top-2.
// sim = (Ah+Al) @ (Bh+Bl)^T ~= Ah@Bh^T + Ah@Bl^T + Al@Bh^T   (Al@Bl^T ~ 2^-22, dropped)
// Grid: (MM/128, NN/128, BB), 256 threads = 8 waves.
// Wave w (wr=w>>1 in 0..3, wc=w&1 in 0..1) owns rows[32wr,+32) x cols[64wc,+64)
// = 2x4 subtiles of 16x16; 3 v_wmma_f32_16x16x32_f16 per subtile per K-chunk.
// ---------------------------------------------------------------------------
__global__ __launch_bounds__(256) void gemm_top2_kernel(
    const f16* __restrict__ Ah, const f16* __restrict__ Al,
    const f16* __restrict__ Bh, const f16* __restrict__ Bl,
    float* __restrict__ sim,       // [BB,NN,MM]
    Part* __restrict__ rowPart,    // [BB,NN, MM/128]
    Part* __restrict__ colPart)    // [BB,MM, NN/128]
{
  __shared__ f16 lds[4 * TILE * LDH];   // 147456 B
  f16* sAh = lds;
  f16* sAl = lds + 1 * TILE * LDH;
  f16* sBh = lds + 2 * TILE * LDH;
  f16* sBl = lds + 3 * TILE * LDH;
  float* Ct = (float*)lds;              // reused after barrier (68096 B)

  const int b = blockIdx.z;
  const int tileN = blockIdx.y, tileM = blockIdx.x;
  const int nbase = tileN * TILE, mbase = tileM * TILE;
  const int tid = threadIdx.x;

  const f16* gAh = Ah + ((size_t)b * NN + nbase) * DD;
  const f16* gAl = Al + ((size_t)b * NN + nbase) * DD;
  const f16* gBh = Bh + ((size_t)b * MM + mbase) * DD;
  const f16* gBl = Bl + ((size_t)b * MM + mbase) * DD;

  // Stage 4 tiles of 128x128 f16. B tiles are staged in the same layout as A (no
  // chunk swap), so B-fragment ds reads use
  // the same conflict-free offsets {4h, 8+4h} dwords as A fragments.
  #pragma unroll
  for (int i = 0; i < 8; ++i) {
    const int idx = tid + 256 * i;      // 0..2047 chunk slots (16 B each)
    const int r = idx >> 4;             // row 0..127
    const int ch = idx & 15;            // 8-half chunk 0..15
    const int j = ch & 3;
    const int jm = j;   // no chunk swap: B staged in same layout as A
    const int chB = (ch & ~3) | jm;
    const float4 ah = *(const float4*)(gAh + (size_t)r * DD + ch * 8);
    const float4 al = *(const float4*)(gAl + (size_t)r * DD + ch * 8);
    const float4 bh = *(const float4*)(gBh + (size_t)r * DD + ch * 8);
    const float4 bl = *(const float4*)(gBl + (size_t)r * DD + ch * 8);
    *(float4*)(sAh + r * LDH + ch * 8)  = ah;
    *(float4*)(sAl + r * LDH + ch * 8)  = al;
    *(float4*)(sBh + r * LDH + chB * 8) = bh;
    *(float4*)(sBl + r * LDH + chB * 8) = bl;
  }
  __syncthreads();

  const int w = tid >> 5, l = tid & 31;
  const int wr = w >> 1, wc = w & 1;
  const int lrow = l & 15;
  const int half = l >> 4;
  const int o1 = 8 * half;        // halfs: first 16B piece of fragment
  const int o2 = 16 + 8 * half;   // halfs: second 16B piece

  v8f acc[2][4] = {};

  for (int kc = 0; kc < 4; ++kc) {
    const int k0 = kc * 32;
    v16h fah[2], fal[2], fbh[4], fbl[4];
    #pragma unroll
    for (int sr = 0; sr < 2; ++sr) {
      const f16* rp = sAh + (wr * 32 + sr * 16 + lrow) * LDH + k0;
      const f16* rq = sAl + (wr * 32 + sr * 16 + lrow) * LDH + k0;
      v8h a0 = *(const v8h*)(rp + o1), a1 = *(const v8h*)(rp + o2);
      v8h b0 = *(const v8h*)(rq + o1), b1 = *(const v8h*)(rq + o2);
      fah[sr] = __builtin_shufflevector(a0, a1, 0,1,2,3,4,5,6,7,8,9,10,11,12,13,14,15);
      fal[sr] = __builtin_shufflevector(b0, b1, 0,1,2,3,4,5,6,7,8,9,10,11,12,13,14,15);
    }
    #pragma unroll
    for (int sc = 0; sc < 4; ++sc) {
      const f16* rp = sBh + (wc * 64 + sc * 16 + lrow) * LDH + k0;
      const f16* rq = sBl + (wc * 64 + sc * 16 + lrow) * LDH + k0;
      v8h a0 = *(const v8h*)(rp + o1), a1 = *(const v8h*)(rp + o2);
      v8h b0 = *(const v8h*)(rq + o1), b1 = *(const v8h*)(rq + o2);
      fbh[sc] = __builtin_shufflevector(a0, a1, 0,1,2,3,4,5,6,7,8,9,10,11,12,13,14,15);
      fbl[sc] = __builtin_shufflevector(b0, b1, 0,1,2,3,4,5,6,7,8,9,10,11,12,13,14,15);
    }
    #pragma unroll
    for (int sr = 0; sr < 2; ++sr)
      #pragma unroll
      for (int sc = 0; sc < 4; ++sc) {
        acc[sr][sc] = __builtin_amdgcn_wmma_f32_16x16x32_f16(
            false, fah[sr], false, fbh[sc], (short)0, acc[sr][sc], false, false);
        acc[sr][sc] = __builtin_amdgcn_wmma_f32_16x16x32_f16(
            false, fah[sr], false, fbl[sc], (short)0, acc[sr][sc], false, false);
        acc[sr][sc] = __builtin_amdgcn_wmma_f32_16x16x32_f16(
            false, fal[sr], false, fbh[sc], (short)0, acc[sr][sc], false, false);
      }
  }

  __syncthreads();  // done reading staged tiles before Ct overwrite

  // Write C tile to global sim and stage into LDS for top-2.
  float* simb = sim + (size_t)b * NN * MM;
  const int rhalf = half << 3;   // +8 rows for high half-wave (C layout)
  #pragma unroll
  for (int sr = 0; sr < 2; ++sr)
    #pragma unroll
    for (int sc = 0; sc < 4; ++sc)
      #pragma unroll
      for (int v = 0; v < 8; ++v) {
        const int r = wr * 32 + sr * 16 + v + rhalf;
        const int c = wc * 64 + sc * 16 + lrow;
        const float val = acc[sr][sc][v];
        simb[(size_t)(nbase + r) * MM + (mbase + c)] = val;
        Ct[r * LDC + c] = val;
      }
  __syncthreads();

  // Per-tile top-2 partials. Strict '>' scan in increasing index order matches
  // top_k tie semantics (equal values ordered by lower index).
  const float NEG = -3.0e38f;
  if (tid < TILE) {
    const int r = tid;
    float v0 = NEG, v1 = NEG; int i0 = 0, i1 = 0;
    for (int c = 0; c < TILE; ++c) {
      const float v = Ct[r * LDC + c];
      if (v > v0)      { v1 = v0; i1 = i0; v0 = v; i0 = c; }
      else if (v > v1) { v1 = v; i1 = c; }
    }
    Part p; p.v0 = v0; p.v1 = v1; p.i0 = mbase + i0; p.i1 = mbase + i1;
    rowPart[((size_t)b * NN + nbase + r) * (MM / TILE) + tileM] = p;
  } else {
    const int c = tid - TILE;
    float v0 = NEG, v1 = NEG; int i0 = 0, i1 = 0;
    for (int r = 0; r < TILE; ++r) {
      const float v = Ct[r * LDC + c];
      if (v > v0)      { v1 = v0; i1 = i0; v0 = v; i0 = r; }
      else if (v > v1) { v1 = v; i1 = r; }
    }
    Part p; p.v0 = v0; p.v1 = v1; p.i0 = nbase + i0; p.i1 = nbase + i1;
    colPart[((size_t)b * MM + mbase + c) * (NN / TILE) + tileN] = p;
  }
}

// ---------------------------------------------------------------------------
// Merge 32 tile-partials per row/col, apply ratio + distance test.
// ---------------------------------------------------------------------------
__global__ __launch_bounds__(256) void reduce_top2_kernel(
    const Part* __restrict__ rowPart, const Part* __restrict__ colPart,
    int* __restrict__ m0raw, int* __restrict__ m1raw)
{
  const int gid = blockIdx.x * 256 + threadIdx.x;      // 0 .. 2*BB*NN-1
  const bool isRow = gid < BB * NN;
  const int idx = isRow ? gid : gid - BB * NN;
  const Part* src = (isRow ? rowPart : colPart) + (size_t)idx * 32;

  float v0 = -3.0e38f, v1 = -3.0e38f; int i0 = -1, i1 = -1;
  for (int t = 0; t < 32; ++t) {
    const Part p = src[t];
    if (p.v0 > v0)      { v1 = v0; i1 = i0; v0 = p.v0; i0 = p.i0; }
    else if (p.v0 > v1) { v1 = p.v0; i1 = p.i0; }
    if (p.v1 > v0)      { v1 = v0; i1 = i0; v0 = p.v1; i0 = p.i1; }
    else if (p.v1 > v1) { v1 = p.v1; i1 = p.i1; }
  }
  const float R2 = (float)(0.8 * 0.8);   // ratio_thresh^2
  const float D2 = (float)(0.7 * 0.7);   // distance_thresh^2
  const float d0 = 2.0f * (1.0f - v0);
  const float d1 = 2.0f * (1.0f - v1);
  const bool ok = (d0 <= R2 * d1) && (d0 <= D2);
  (isRow ? m0raw : m1raw)[idx] = ok ? i0 : -1;
}

// ---------------------------------------------------------------------------
// Mutual check + emit matches (as float) and mscores.
// d_out: matches0[BN] | matches1[BM] | mscores0[BN] | mscores1[BM] | sim
// ---------------------------------------------------------------------------
__global__ __launch_bounds__(256) void finalize_kernel(
    const int* __restrict__ m0raw, const int* __restrict__ m1raw,
    float* __restrict__ out)
{
  const int gid = blockIdx.x * 256 + threadIdx.x;      // 0 .. BB*NN-1 (NN==MM)
  if (gid >= BB * NN) return;
  const int b = gid / NN, i = gid % NN;

  const int m0 = m0raw[gid];
  const bool ok0 = (m0 > -1) && (m1raw[b * MM + m0] == i);
  out[gid]               = ok0 ? (float)m0 : -1.0f;   // matches0
  out[2 * BB * NN + gid] = ok0 ? 1.0f : 0.0f;         // mscores0

  const int m1 = m1raw[gid];
  const bool ok1 = (m1 > -1) && (m0raw[b * NN + m1] == i);
  out[BB * NN + gid]     = ok1 ? (float)m1 : -1.0f;   // matches1
  out[3 * BB * NN + gid] = ok1 ? 1.0f : 0.0f;         // mscores1
}

// ---------------------------------------------------------------------------
extern "C" void kernel_launch(void* const* d_in, const int* in_sizes, int n_in,
                              void* d_out, int out_size, void* d_ws, size_t ws_size,
                              hipStream_t stream) {
  const float* d0 = (const float*)d_in[0];   // [BB,NN,DD] f32
  const float* d1 = (const float*)d_in[1];   // [BB,MM,DD] f32
  float* out = (float*)d_out;

  // Workspace layout (~64.3 MiB)
  char* ws = (char*)d_ws;
  f16* d0h = (f16*)ws;                                      // BB*NN*DD f16
  f16* d0l = d0h + (size_t)BB * NN * DD;
  f16* d1h = d0l + (size_t)BB * NN * DD;
  f16* d1l = d1h + (size_t)BB * MM * DD;
  Part* rowPart = (Part*)(d1l + (size_t)BB * MM * DD);      // BB*NN*32
  Part* colPart = rowPart + (size_t)BB * NN * (MM / TILE);  // BB*MM*32
  int* m0raw = (int*)(colPart + (size_t)BB * MM * (NN / TILE));
  int* m1raw = m0raw + BB * NN;

  float* sim = out + 4 * BB * NN;  // sim output region

  l2norm_split_kernel<<<BB * NN / 8, 256, 0, stream>>>(d0, d0h, d0l, BB * NN);
  l2norm_split_kernel<<<BB * MM / 8, 256, 0, stream>>>(d1, d1h, d1l, BB * MM);

  gemm_top2_kernel<<<dim3(MM / TILE, NN / TILE, BB), 256, 0, stream>>>(
      d0h, d0l, d1h, d1l, sim, rowPart, colPart);

  reduce_top2_kernel<<<(2 * BB * NN) / 256, 256, 0, stream>>>(
      rowPart, colPart, m0raw, m1raw);

  finalize_kernel<<<(BB * NN) / 256, 256, 0, stream>>>(m0raw, m1raw, out);
}
